// MambaBlock_49572512530867
// MI455X (gfx1250) — matrix operand order repaired and checked by reading, not compiled, not run on hardware
//
#include <hip/hip_runtime.h>
#include <hip/hip_bf16.h>

// ---- problem constants ----
#define D_MODEL 768
#define D_INNER 1536
#define NCOLS   (2 * D_INNER)     // 3072
#define DSTATE  16
#define DTRANK  48
#define XDBL_N  80                // DT_RANK + 2*D_STATE
#define BATCH   8
#define SEQ     2048
#define ROWS    (BATCH * SEQ)     // 16384

typedef __attribute__((ext_vector_type(16))) __bf16 v16bf;
typedef __attribute__((ext_vector_type(8)))  float  v8f;

__device__ __forceinline__ __bf16 to_bf16(float f) { return (__bf16)f; }

// ---- CDNA5 async global->LDS copy (ASYNCcnt-tracked), 16B per lane ----
// Builtin expects (AS1 int4*, AS3 int4*, imm offset, imm cpol).
typedef int v4i __attribute__((vector_size(16)));
typedef __attribute__((address_space(1))) v4i gv4i_t;
typedef __attribute__((address_space(3))) v4i lv4i_t;

__device__ __forceinline__ void async_copy_b128(const float* g, float* l) {
#if __has_builtin(__builtin_amdgcn_global_load_async_to_lds_b128)
    __builtin_amdgcn_global_load_async_to_lds_b128(
        (gv4i_t*)g, (lv4i_t*)l, 0, 0);
#else
    const unsigned ldsoff = (unsigned)(unsigned long long)(uintptr_t)l;
    asm volatile("global_load_async_to_lds_b128 %0, %1, off"
                 :: "v"(ldsoff), "v"((unsigned long long)(uintptr_t)g)
                 : "memory");
#endif
}

__device__ __forceinline__ void wait_async0() {
#if __has_builtin(__builtin_amdgcn_s_wait_asynccnt)
    __builtin_amdgcn_s_wait_asynccnt(0);
#else
    asm volatile("s_wait_asynccnt 0x0" ::: "memory");
#endif
}

// =====================================================================
// Kernel 1: xz = H(16384x768) @ W_in(768x3072), split into x_raw / res.
// Block tile 128(M) x 64(N), K-step 32, 256 threads = 8 waves (4x2).
// A tile staged via GLOBAL_LOAD_ASYNC_TO_LDS_B128 (f32, padded stride 36,
// converted to bf16 at fragment build); B tile staged transposed as bf16.
// Each wave: 2x2 tiles of v_wmma_f32_16x16x32_bf16.
// =====================================================================
__global__ __launch_bounds__(256)
void k_gemm_in(const float* __restrict__ H, const float* __restrict__ W,
               float* __restrict__ xraw, float* __restrict__ resb)
{
    __shared__ float  lAf[128][36];  // [m][k] f32, pad 32->36 (36-bank stride)
    __shared__ __bf16 lB[64][40];    // [n][k] bf16 (transposed B tile)

    const int tid  = threadIdx.x;
    const int lane = tid & 31;
    const int wave = tid >> 5;
    const int wm   = wave & 3;       // wave row (0..3) -> 32 rows each
    const int wn   = wave >> 2;      // wave col (0..1) -> 32 cols each
    const int rb   = blockIdx.x * 128;
    const int cb   = blockIdx.y * 64;

    v8f acc[2][2];
    #pragma unroll
    for (int i = 0; i < 2; ++i)
        #pragma unroll
        for (int j = 0; j < 2; ++j)
            #pragma unroll
            for (int r = 0; r < 8; ++r) acc[i][j][r] = 0.0f;

    const int l15 = lane & 15;
    const int ahk = (lane < 16) ? 0 : 8;    // A fragment K base
    const int bhk = (lane < 16) ? 0 : 8;    // B fragment K base

    for (int kb = 0; kb < D_MODEL; kb += 32) {
        // ---- A tile: 128x32 f32 via async DMA to LDS (4 x b128 per thread)
        {
            const int col  = (tid & 7) * 4;      // 0..28
            const int row0 = tid >> 3;           // 0..31
            #pragma unroll
            for (int p = 0; p < 4; ++p) {
                const int row = row0 + p * 32;
                async_copy_b128(H + (size_t)(rb + row) * D_MODEL + kb + col,
                                &lAf[row][col]);
            }
        }
        // ---- B tile transposed: 32x64 f32 -> lB[n][k] bf16 (manual, overlaps async)
        {
            const int col  = (tid & 15) * 4;     // n offset 0..60
            const int row0 = tid >> 4;           // k 0..15
            #pragma unroll
            for (int p = 0; p < 2; ++p) {
                const int k = row0 + p * 16;
                const float4 v = *(const float4*)(W + (size_t)(kb + k) * NCOLS + cb + col);
                lB[col + 0][k] = to_bf16(v.x);
                lB[col + 1][k] = to_bf16(v.y);
                lB[col + 2][k] = to_bf16(v.z);
                lB[col + 3][k] = to_bf16(v.w);
            }
        }
        wait_async0();
        __syncthreads();

        #pragma unroll
        for (int ti = 0; ti < 2; ++ti) {
            const int arow = wm * 32 + ti * 16 + l15;
            const float* pa = &lAf[arow][ahk];
            v16bf a;
            #pragma unroll
            for (int e = 0; e < 8; ++e) {
                a[e]     = to_bf16(pa[e]);
                a[e + 8] = to_bf16(pa[e + 16]);
            }
            #pragma unroll
            for (int tj = 0; tj < 2; ++tj) {
                const int bn = wn * 32 + tj * 16 + l15;
                const __bf16* pb = &lB[bn][bhk];
                v16bf b;
                #pragma unroll
                for (int e = 0; e < 16; ++e) b[e] = pb[e + (e & 8)];
                acc[ti][tj] = __builtin_amdgcn_wmma_f32_16x16x32_bf16(
                    false, a, false, b, (short)0, acc[ti][tj], false, false);
            }
        }
        __syncthreads();
    }

    // ---- epilogue: uniform per-block split (cb multiple of 64; 1536%64==0)
    float* dst;
    int basecol;
    if (cb < D_INNER) { dst = xraw; basecol = cb; }
    else              { dst = resb; basecol = cb - D_INNER; }

    const int roff = (lane < 16) ? 0 : 8;
    #pragma unroll
    for (int ti = 0; ti < 2; ++ti) {
        #pragma unroll
        for (int tj = 0; tj < 2; ++tj) {
            const int gcol = basecol + wn * 32 + tj * 16 + l15;
            #pragma unroll
            for (int r = 0; r < 8; ++r) {
                const int grow = rb + wm * 32 + ti * 16 + r + roff;
                dst[(size_t)grow * D_INNER + gcol] = acc[ti][tj][r];
            }
        }
    }
}

// =====================================================================
// Kernel 2: depthwise causal conv (K=4) + bias + SiLU
// =====================================================================
__global__ __launch_bounds__(256)
void k_conv_silu(const float* __restrict__ xraw, const float* __restrict__ cw,
                 const float* __restrict__ cbias, float* __restrict__ xact)
{
    const size_t idx = (size_t)blockIdx.x * 256 + threadIdx.x;
    const int d = (int)(idx % D_INNER);
    const size_t bl = idx / D_INNER;
    const int l = (int)(bl % SEQ);
    const size_t b = bl / SEQ;

    float acc = cbias[d];
    #pragma unroll
    for (int k = 0; k < 4; ++k) {
        const int ll = l + k - 3;
        if (ll >= 0)
            acc += cw[d * 4 + k] * xraw[((b * SEQ) + ll) * D_INNER + d];
    }
    xact[idx] = acc / (1.0f + __expf(-acc));
}

// =====================================================================
// Kernel 3: x_dbl = x_act(16384x1536) @ W_x^T (W_x is 80x1536 row-major)
// 4 waves / block, each wave one 16x16 tile; K loop 1536/32.
// Fragments gathered straight from global (contiguous K both operands).
// =====================================================================
__global__ __launch_bounds__(128)
void k_xdbl(const float* __restrict__ xact, const float* __restrict__ Wx,
            float* __restrict__ xdbl)
{
    const int lane = threadIdx.x & 31;
    const int wave = threadIdx.x >> 5;
    const int rb = (blockIdx.x * 4 + wave) * 16;
    const int nb = blockIdx.y * 16;
    const int l15 = lane & 15;
    const int ahk = (lane < 16) ? 0 : 8;
    const int bhk = (lane < 16) ? 0 : 8;

    v8f acc;
    #pragma unroll
    for (int r = 0; r < 8; ++r) acc[r] = 0.0f;

    const float* parow = xact + (size_t)(rb + l15) * D_INNER + ahk;
    const float* pbrow = Wx   + (size_t)(nb + l15) * D_INNER + bhk;  // W_x[n][k]

    for (int kb = 0; kb < D_INNER; kb += 32) {
        const float* pa = parow + kb;
        v16bf a;
        #pragma unroll
        for (int e = 0; e < 8; ++e) { a[e] = to_bf16(pa[e]); a[e + 8] = to_bf16(pa[e + 16]); }
        const float* pb = pbrow + kb;
        v16bf b;
        #pragma unroll
        for (int e = 0; e < 16; ++e) b[e] = to_bf16(pb[e + (e & 8)]);
        acc = __builtin_amdgcn_wmma_f32_16x16x32_bf16(
            false, a, false, b, (short)0, acc, false, false);
    }

    const int roff = (lane < 16) ? 0 : 8;
    #pragma unroll
    for (int r = 0; r < 8; ++r)
        xdbl[(size_t)(rb + r + roff) * XDBL_N + nb + l15] = acc[r];
}

// =====================================================================
// Kernel 4: dt = softplus(x_dbl[:, :48] @ W_dt^T + b_dt)
// K=48 padded to 64 (two WMMA K-steps; second half zero-padded).
// =====================================================================
__global__ __launch_bounds__(128)
void k_dtproj(const float* __restrict__ xdbl, const float* __restrict__ Wdt,
              const float* __restrict__ bdt, float* __restrict__ dt)
{
    const int lane = threadIdx.x & 31;
    const int wave = threadIdx.x >> 5;
    const int rb = (blockIdx.x * 4 + wave) * 16;
    const int nb = blockIdx.y * 16;
    const int l15 = lane & 15;
    const int ahk = (lane < 16) ? 0 : 8;
    const int bhk = (lane < 16) ? 0 : 8;

    v8f acc;
    #pragma unroll
    for (int r = 0; r < 8; ++r) acc[r] = 0.0f;

    const float* parow = xdbl + (size_t)(rb + l15) * XDBL_N + ahk;
    const float* pbrow = Wdt  + (size_t)(nb + l15) * DTRANK + bhk;

    // kb = 0 : all K in [0,32) valid (<48)
    {
        v16bf a, b;
        #pragma unroll
        for (int e = 0; e < 8; ++e) { a[e] = to_bf16(parow[e]); a[e + 8] = to_bf16(parow[e + 16]); }
        #pragma unroll
        for (int e = 0; e < 16; ++e) b[e] = to_bf16(pbrow[e + (e & 8)]);
        acc = __builtin_amdgcn_wmma_f32_16x16x32_bf16(
            false, a, false, b, (short)0, acc, false, false);
    }
    // kb = 32 : K in [32,64); K >= 48 zero-padded
    {
        const float* pa = parow + 32;
        v16bf a;
        #pragma unroll
        for (int e = 0; e < 8; ++e) a[e] = to_bf16(pa[e]);       // K = 32+ahk+e <= 47
        #pragma unroll
        for (int e = 8; e < 16; ++e) a[e] = to_bf16(0.0f);        // K = 48..63 pad
        const float* pb = pbrow + 32;
        v16bf b;
        #pragma unroll
        for (int e = 0; e < 16; ++e) {
            const int k = 32 + bhk + e + (e & 8);
            __bf16 t = to_bf16(0.0f);
            if (k < DTRANK) t = to_bf16(pb[e + (e & 8)]);
            b[e] = t;
        }
        acc = __builtin_amdgcn_wmma_f32_16x16x32_bf16(
            false, a, false, b, (short)0, acc, false, false);
    }

    const float bias = bdt[nb + l15];
    const int roff = (lane < 16) ? 0 : 8;
    #pragma unroll
    for (int r = 0; r < 8; ++r) {
        const float v = acc[r] + bias;
        const float sp = (v > 20.0f) ? v : __logf(1.0f + __expf(v));
        dt[(size_t)(rb + r + roff) * D_INNER + nb + l15] = sp;
    }
}

// =====================================================================
// Kernel 5: selective scan + epilogue.
// 16 lanes of a half-wave carry the 16 state channels of one (b,d);
// per-step cross-lane reduction via __shfl_xor for y = sum_s h_s * C_s.
// Streaming rows prefetched ahead (global_prefetch_b8).
// out = (y + x*D) * silu(res)
// =====================================================================
__global__ __launch_bounds__(256)
void k_scan(const float* __restrict__ xact, const float* __restrict__ dt,
            const float* __restrict__ xdbl, const float* __restrict__ resb,
            const float* __restrict__ Alog, const float* __restrict__ Dp,
            float* __restrict__ out)
{
    const int t = blockIdx.x * 256 + threadIdx.x;
    const int s = t & 15;            // state channel
    const int g = t >> 4;            // (b,d) group
    const int d = g % D_INNER;
    const int b = g / D_INNER;

    const float a_s = -__expf(Alog[d * DSTATE + s]);
    const float Dv  = Dp[d];
    float h = 0.0f;
    const size_t base = (size_t)b * SEQ;

    for (int l = 0; l < SEQ; ++l) {
        const size_t bl = base + l;
        const float dtv = dt[bl * D_INNER + d];
        const float xv  = xact[bl * D_INNER + d];
        const float Bv  = xdbl[bl * XDBL_N + DTRANK + s];
        const float Cv  = xdbl[bl * XDBL_N + DTRANK + DSTATE + s];

        // prefetch 16 steps ahead (speculative; dropped if OOB)
        __builtin_prefetch(&dt[(bl + 16) * D_INNER + d], 0, 1);
        __builtin_prefetch(&xact[(bl + 16) * D_INNER + d], 0, 1);

        h = __expf(dtv * a_s) * h + (dtv * xv) * Bv;
        float y = h * Cv;
        y += __shfl_xor(y, 1, 32);
        y += __shfl_xor(y, 2, 32);
        y += __shfl_xor(y, 4, 32);
        y += __shfl_xor(y, 8, 32);
        if (s == 0) {
            const float r = resb[bl * D_INNER + d];
            out[bl * D_INNER + d] = (y + xv * Dv) * (r / (1.0f + __expf(-r)));
        }
    }
}

// =====================================================================
extern "C" void kernel_launch(void* const* d_in, const int* in_sizes, int n_in,
                              void* d_out, int out_size, void* d_ws, size_t ws_size,
                              hipStream_t stream) {
    const float* H    = (const float*)d_in[0];  // hidden_states (8,2048,768)
    const float* Win  = (const float*)d_in[1];  // W_in (768,3072)
    const float* cw   = (const float*)d_in[2];  // conv_w (1536,1,4)
    const float* cb   = (const float*)d_in[3];  // conv_b (1536)
    const float* Wx   = (const float*)d_in[4];  // W_x (80,1536)
    const float* Wdt  = (const float*)d_in[5];  // W_dt (1536,48)
    const float* bdt  = (const float*)d_in[6];  // b_dt (1536)
    const float* Alog = (const float*)d_in[7];  // A_log (1536,16)
    const float* Dp   = (const float*)d_in[8];  // D_param (1536)
    float* out = (float*)d_out;

    float* ws = (float*)d_ws;
    const size_t NX = (size_t)ROWS * D_INNER;   // 25,165,824 floats
    float* xraw = ws;                // raw x half of xz; later reused for dt
    float* resb = ws + NX;           // res half
    float* xact = ws + 2 * NX;       // conv+silu output
    float* xdbl = ws + 3 * NX;       // (16384 x 80)
    float* dtb  = xraw;              // xraw dead after conv -> reuse for dt

    k_gemm_in<<<dim3(ROWS / 128, NCOLS / 64), 256, 0, stream>>>(H, Win, xraw, resb);
    k_conv_silu<<<(unsigned)(NX / 256), 256, 0, stream>>>(xraw, cw, cb, xact);
    k_xdbl<<<dim3(ROWS / 64, XDBL_N / 16), 128, 0, stream>>>(xact, Wx, xdbl);
    k_dtproj<<<dim3(ROWS / 64, D_INNER / 16), 128, 0, stream>>>(xdbl, Wdt, bdt, dtb);
    k_scan<<<(BATCH * D_INNER * 16) / 256, 256, 0, stream>>>(xact, dtb, xdbl, resb, Alog, Dp, out);
}
